// SemanticLoss_70738111365281
// MI455X (gfx1250) — compile-verified, matrix operand order checked by reading, not run on hardware
//
#include <hip/hip_runtime.h>
#include <hip/hip_bf16.h>

// ---------------------------------------------------------------------------
// Fused STEGO + depth-guided correlation loss for MI455X (gfx1250, wave32).
// v4: B tiles staged into LDS by the Tensor Data Mover (6-arg
//     tensor_load_to_lds builtin, double-buffered, TENSORcnt-synchronized);
//     WMMA feeds from ds_load_b128. 4 waves per workgroup split the column
//     stream; per-wave LDS regions, single cross-wave barrier at the end.
// ---------------------------------------------------------------------------

#define BATCH   2
#define NPTS    4096
#define DF      384      // DINO feature dim (multiple of 32)
#define DC      90       // code dim
#define DCP     96       // code dim padded to multiple of 32
#define KNN     7
#define INV_T   10.0f    // 1 / temperature (0.1)
#define NEG_HALF_INV_SIG2 (-2.0f)  // -1/(2*sigma^2), sigma = 0.5
#define L2EPS   1e-8f
#define NWAVE   4        // waves per workgroup (one per SIMD32 of the WGP)
#define LGMAX   10.0f    // fixed softmax max: logits = 10*cos <= ~10

typedef __attribute__((ext_vector_type(16))) _Float16 v16h;
typedef __attribute__((ext_vector_type(8)))  _Float16 v8h;
typedef __attribute__((ext_vector_type(8)))  float    v8f;
typedef __attribute__((ext_vector_type(4)))  unsigned int u32x4;
typedef __attribute__((ext_vector_type(8)))  int          i32x8;
typedef __attribute__((ext_vector_type(4)))  int          i32x4;

// ---- Tensor Data Mover: async 2D tile load (Global -> LDS), D# per ISA §8.
//   rowlen  = tile/tensor dim0 and dim0 stride (in 2-byte elements)
//   nrows   = tile dim1
__device__ __forceinline__ void tdm_load_tile(void* lds_dst, const _Float16* gsrc,
                                              int rowlen, int nrows) {
  unsigned long long ga = (unsigned long long)(uintptr_t)gsrc;
  unsigned lds_off = (unsigned)(uintptr_t)lds_dst;   // flat->LDS: addr[31:0]

  u32x4 g0;
  g0.x = 1u;                                          // count=1, user descriptor
  g0.y = lds_off;                                     // LDS byte address
  g0.z = (unsigned)(ga & 0xFFFFFFFFu);                // global_addr[31:0]
  g0.w = (unsigned)((ga >> 32) & 0x1FFFFFFu)          // global_addr[56:32]
       | (2u << 30);                                  // type = 2 ("image")

  i32x8 g1;
  g1[0] = (1 << 16);                                  // data_size = 1 (2 bytes)
  g1[1] = (rowlen & 0xFFFF) << 16;                    // tensor_dim0[15:0] @ bits 63:48
  g1[2] = ((rowlen >> 16) & 0xFFFF)                   // tensor_dim0[31:16]
        | ((NPTS & 0xFFFF) << 16);                    // tensor_dim1[15:0]
  g1[3] = ((NPTS >> 16) & 0xFFFF)                     // tensor_dim1[31:16]
        | ((rowlen & 0xFFFF) << 16);                  // tile_dim0
  g1[4] = nrows & 0xFFFF;                             // tile_dim1 (tile_dim2 = 0)
  g1[5] = rowlen;                                     // tensor_dim0_stride[31:0]
  g1[6] = 0;                                          // stride hi / dim1_stride lo
  g1[7] = 0;

  i32x4 gz4 = {0, 0, 0, 0};                           // groups 2/3: 2D tile
  i32x8 gz8 = {0, 0, 0, 0, 0, 0, 0, 0};
  asm volatile("" ::: "memory");                      // don't sink loads past issue
  __builtin_amdgcn_tensor_load_to_lds(g0, g1, gz4, gz4, gz8, 0);
}

template <int N>
__device__ __forceinline__ void tdm_wait() {
  __builtin_amdgcn_s_wait_tensorcnt(N);               // constant operand required
  asm volatile("" ::: "memory");                      // don't hoist loads above wait
}

// Read one 16x32 f16 B chunk from a row-major staged LDS tile in the CDNA5
// WMMA register layout:
//   lanes 0-15 : N = lane,     halves = K+0..7  and K+16..23
//   lanes 16-31: N = lane-16,  halves = K+8..15 and K+24..31
__device__ __forceinline__ v16h lds_tile_chunk(const _Float16* st, int ld, int kk,
                                               int lane) {
  int r  = lane & 15;
  int k0 = kk + ((lane & 16) ? 8 : 0);
  const _Float16* p = st + r * ld + k0;
  union { v16h v; v8h h[2]; } u;
  u.h[0] = *(const v8h*)(p);
  u.h[1] = *(const v8h*)(p + 16);
  return u.v;
}

// Same layout, loaded from global memory (used once for the resident A tiles).
__device__ __forceinline__ v16h load_tile_chunk(const _Float16* __restrict__ base,
                                                int ld, int row0, int kk, int lane) {
  int r  = row0 + (lane & 15);
  int k0 = kk + ((lane & 16) ? 8 : 0);
  const _Float16* p = base + (size_t)r * ld + k0;
  union { v16h v; v8h h[2]; } u;
  u.h[0] = *(const v8h*)(p);
  u.h[1] = *(const v8h*)(p + 16);
  return u.v;
}

// Sorted (ascending) top-7 insertion; tv[0] is the current minimum.
__device__ __forceinline__ void topk_insert(float (&tv)[KNN], int (&ti)[KNN],
                                            float v, int idx) {
  if (v > tv[0]) {
    tv[0] = v; ti[0] = idx;
#pragma unroll
    for (int j = 0; j < KNN - 1; ++j) {
      bool sw = tv[j] > tv[j + 1];
      float a = sw ? tv[j + 1] : tv[j];
      float b = sw ? tv[j]     : tv[j + 1];
      int   c = sw ? ti[j + 1] : ti[j];
      int   d = sw ? ti[j]     : ti[j + 1];
      tv[j] = a; tv[j + 1] = b; ti[j] = c; ti[j + 1] = d;
    }
  }
}

// --------------------------- prep: l2-normalize ----------------------------
__global__ __launch_bounds__(256)
void prep_normalize(const float* __restrict__ codes, const float* __restrict__ feats,
                    _Float16* __restrict__ ch, _Float16* __restrict__ fh) {
  int gtid = blockIdx.x * blockDim.x + threadIdx.x;
  int row  = gtid >> 5;
  int lane = threadIdx.x & 31;
  if (row >= BATCH * NPTS) return;

  const float* fr = feats + (size_t)row * DF;
  float s = 0.f;
  for (int k = lane; k < DF; k += 32) { float v = fr[k]; s += v * v; }
#pragma unroll
  for (int off = 16; off; off >>= 1) s += __shfl_xor(s, off);
  float invf = rsqrtf(s + L2EPS);

  const float* cr = codes + (size_t)row * DC;
  float s2 = 0.f;
  for (int k = lane; k < DC; k += 32) { float v = cr[k]; s2 += v * v; }
#pragma unroll
  for (int off = 16; off; off >>= 1) s2 += __shfl_xor(s2, off);
  float invc = rsqrtf(s2 + L2EPS);

  _Float16* fo = fh + (size_t)row * DF;
  for (int k = lane; k < DF; k += 32) fo[k] = (_Float16)(fr[k] * invf);
  _Float16* co = ch + (size_t)row * DCP;
  for (int k = lane; k < DCP; k += 32)
    co[k] = (_Float16)((k < DC) ? cr[k] * invc : 0.f);
}

// --------------------------- fused main kernel -----------------------------
__global__ __launch_bounds__(32 * NWAVE)
void stego_fused(const _Float16* __restrict__ fh, const _Float16* __restrict__ ch,
                 const float* __restrict__ depth, float* __restrict__ acc) {
  const int tileid = blockIdx.x;            // 0 .. BATCH*(NPTS/16)-1
  const int b      = tileid >> 8;           // 256 row tiles per batch
  const int r0     = (tileid & 255) << 4;
  const int lane   = threadIdx.x & 31;
  const int wave   = threadIdx.x >> 5;

  const _Float16* fb = fh + (size_t)b * NPTS * DF;
  const _Float16* cb = ch + (size_t)b * NPTS * DCP;
  const float*    db = depth + (size_t)b * NPTS;

  // Per-wave LDS: double-buffered TDM staging for the B tiles + reshard tiles.
  __shared__ _Float16 stF[NWAVE][2][16 * DF];   // 4 x 2 x 12 KB
  __shared__ _Float16 stC[NWAVE][2][16 * DCP];  // 4 x 2 x  3 KB
  __shared__ float    sf[NWAVE][256];
  __shared__ float    sc[NWAVE][256];
  __shared__ float    mv[NWAVE][16][8];         // top-7 values + softmax sum
  __shared__ int      mi[NWAVE][16][KNN];       // top-7 indices

  // Resident A tiles (this workgroup's 16 rows; replicated per wave).
  v16h af[DF / 32];
#pragma unroll
  for (int q = 0; q < DF / 32; ++q) af[q] = load_tile_chunk(fb, DF, r0, q * 32, lane);
  v16h ac[DCP / 32];
#pragma unroll
  for (int q = 0; q < DCP / 32; ++q) ac[q] = load_tile_chunk(cb, DCP, r0, q * 32, lane);

  // After the LDS reshard each lane owns row = lane>>1, cols (lane&1)*8..+7.
  const int sub = lane & 1;
  const int row = lane >> 1;
  const float d_row = db[r0 + row];

  float tv[KNN]; int ti[KNN];
#pragma unroll
  for (int j = 0; j < KNN; ++j) { tv[j] = -1e30f; ti[j] = 0; }
  float ssum = 0.f, dep_acc = 0.f;

  const int cstore = lane & 15;             // C/D layout: VGPR i -> row rbase+i
  const int rbase  = (lane >> 4) << 3;

  const int t0 = wave * (NPTS / 16 / NWAVE);
  const int t1 = t0 + (NPTS / 16 / NWAVE);

  // Prime the pipeline: TDM both B tiles of the first column tile.
  tdm_load_tile(&stF[wave][0][0], fb + (size_t)(t0 * 16) * DF,  DF,  16);
  tdm_load_tile(&stC[wave][0][0], cb + (size_t)(t0 * 16) * DCP, DCP, 16);

  int p = 0;
  for (int t = t0; t < t1; ++t, p ^= 1) {
    const int c0 = t << 4;

    // Issue next tile's async DMA, then wait only for the current tile
    // (TDM ops from one wave complete in order).
    if (t + 1 < t1) {
      tdm_load_tile(&stF[wave][p ^ 1][0], fb + (size_t)((t + 1) * 16) * DF,  DF,  16);
      tdm_load_tile(&stC[wave][p ^ 1][0], cb + (size_t)((t + 1) * 16) * DCP, DCP, 16);
      tdm_wait<2>();
    } else {
      tdm_wait<0>();
    }

    const _Float16* bF = &stF[wave][p][0];
    const _Float16* bC = &stC[wave][p][0];

    v8f Cf = {};
#pragma unroll
    for (int q = 0; q < DF / 32; ++q) {
      v16h bq = lds_tile_chunk(bF, DF, q * 32, lane);
      Cf = __builtin_amdgcn_wmma_f32_16x16x32_f16(false, af[q], false, bq,
                                                  (short)0, Cf, false, false);
    }
    v8f Cc = {};
#pragma unroll
    for (int q = 0; q < DCP / 32; ++q) {
      v16h bq = lds_tile_chunk(bC, DCP, q * 32, lane);
      Cc = __builtin_amdgcn_wmma_f32_16x16x32_f16(false, ac[q], false, bq,
                                                  (short)0, Cc, false, false);
    }

    // Reshard the 16x16 tiles through this wave's LDS region (wave-local:
    // DS ops from one wave are processed in order, no barrier needed).
#pragma unroll
    for (int i = 0; i < 8; ++i) {
      sf[wave][(rbase + i) * 16 + cstore] = Cf[i];
      sc[wave][(rbase + i) * 16 + cstore] = Cc[i];
    }

    const float4* dp = reinterpret_cast<const float4*>(db + c0 + sub * 8);
    float4 d0 = dp[0], d1 = dp[1];
    float dc[8] = {d0.x, d0.y, d0.z, d0.w, d1.x, d1.y, d1.z, d1.w};

#pragma unroll
    for (int j = 0; j < 8; ++j) {
      int col = sub * 8 + j;
      float simv = sf[wave][row * 16 + col];
      float cosv = sc[wave][row * 16 + col];
      topk_insert(tv, ti, simv, c0 + col);
      ssum += __expf(cosv * INV_T - LGMAX);   // fixed-max streaming softmax
      float dd = d_row - dc[j];
      dep_acc = fmaf(__expf(dd * dd * NEG_HALF_INV_SIG2), 1.0f - cosv, dep_acc);
    }
  }

  // Merge the two lanes of each row within the wave.
  ssum += __shfl_xor(ssum, 1);
#pragma unroll
  for (int j = 0; j < KNN; ++j) {
    float pv = __shfl_xor(tv[j], 1);
    int   pi = __shfl_xor(ti[j], 1);
    topk_insert(tv, ti, pv, pi);
  }

  // Depth partial: one atomic per wave.
#pragma unroll
  for (int off = 16; off; off >>= 1) dep_acc += __shfl_xor(dep_acc, off);
  if (lane == 0) atomicAdd(&acc[1], dep_acc);

  // Publish per-row wave partials (even lane of each pair owns the row).
  if (sub == 0) {
#pragma unroll
    for (int j = 0; j < KNN; ++j) { mv[wave][row][j] = tv[j]; mi[wave][row][j] = ti[j]; }
    mv[wave][row][KNN] = ssum;
  }
  __syncthreads();

  // Wave 0, lanes 0-15: cross-wave merge + positive-logit gather.
  if (wave == 0 && lane < 16) {
    float tv2[KNN]; int ti2[KNN];
#pragma unroll
    for (int j = 0; j < KNN; ++j) { tv2[j] = -1e30f; ti2[j] = 0; }
    float stot = 0.f;
#pragma unroll
    for (int w = 0; w < NWAVE; ++w) {
      stot += mv[w][lane][KNN];
#pragma unroll
      for (int j = 0; j < KNN; ++j) topk_insert(tv2, ti2, mv[w][lane][j], mi[w][lane][j]);
    }
    float lse = LGMAX + __logf(stot);

    const _Float16* crow = cb + (size_t)(r0 + lane) * DCP;
    float psum = 0.f;
#pragma unroll
    for (int j = 0; j < KNN; ++j) {
      const _Float16* ck = cb + (size_t)ti2[j] * DCP;
      float dot = 0.f;
      for (int k = 0; k < DCP; ++k) dot = fmaf((float)crow[k], (float)ck[k], dot);
      psum += dot * INV_T - lse;
    }
    atomicAdd(&acc[0], psum);
  }
}

// ------------------------------ epilogue -----------------------------------
__global__ void zero_acc(float* acc) { acc[0] = 0.f; acc[1] = 0.f; }

__global__ void finalize(const float* __restrict__ acc, float* __restrict__ out) {
  float ls = -acc[0] / ((float)BATCH * NPTS * KNN);
  float ld =  acc[1] / ((float)BATCH * NPTS * (float)NPTS);
  out[0] = ls;
  out[1] = ld;
  out[2] = ls + 0.3f * ld;
}

// ------------------------------ launcher -----------------------------------
extern "C" void kernel_launch(void* const* d_in, const int* in_sizes, int n_in,
                              void* d_out, int out_size, void* d_ws, size_t ws_size,
                              hipStream_t stream) {
  const float* codes = (const float*)d_in[0];   // (B, N, 90)
  const float* feats = (const float*)d_in[1];   // (B, N, 384)
  const float* depth = (const float*)d_in[2];   // (B, N)

  float*     acc = (float*)d_ws;
  _Float16*  fh  = (_Float16*)((char*)d_ws + 256);
  _Float16*  ch  = (_Float16*)((char*)d_ws + 256 + (size_t)BATCH * NPTS * DF * sizeof(_Float16));

  zero_acc<<<1, 1, 0, stream>>>(acc);
  prep_normalize<<<(BATCH * NPTS) / 8, 256, 0, stream>>>(codes, feats, ch, fh);
  stego_fused<<<BATCH * (NPTS / 16), 32 * NWAVE, 0, stream>>>(fh, ch, depth, acc);
  finalize<<<1, 1, 0, stream>>>(acc, (float*)d_out);
}
